// AttModelPerParts_16149077033207
// MI455X (gfx1250) — matrix operand order repaired and checked by reading, not compiled, not run on hardware
//
#include <hip/hip_runtime.h>

// ---------------- CDNA5 WMMA types ----------------
typedef __attribute__((ext_vector_type(16))) __bf16 v16bf;
typedef __attribute__((ext_vector_type(8)))  float  v8f;

#define BM 128
#define BN 128
#define BK 32
#define KSTR 40   // LDS row stride in bf16 elems: 80 bytes -> 16B-aligned rows

#define PI_F 3.14159265358979323846f

// model constants
#define FEAT_C 66
#define NPART  6
#define BSZ    64
#define IN_N   50
#define DCTN   20
#define VLEN   35
#define VWIN   16
#define DMODEL 1024

// dct_m[k][i], N=35 (orthonormal DCT-II; inverse == transpose)
__device__ __forceinline__ float dct_coef(int k, int i) {
    float w = (k == 0) ? sqrtf(1.0f / 35.0f) : sqrtf(2.0f / 35.0f);
    return w * cosf(PI_F * ((float)i + 0.5f) * (float)k / 35.0f);
}

__device__ __forceinline__ unsigned pack_bf16(float a, float b) {
    unsigned short ua = __builtin_bit_cast(unsigned short, (__bf16)a);
    unsigned short ub = __builtin_bit_cast(unsigned short, (__bf16)b);
    return (unsigned)ua | ((unsigned)ub << 16);
}

// ---------------- WMMA fragment gathers from LDS ----------------
// A (16x32 bf16): lane L holds row M=L%16; v<4 -> K=2v+{0,1}(+8 if L>=16),
//                 v>=4 -> K=16+2(v-4)+{0,1}(+8 if L>=16)
__device__ __forceinline__ v16bf load_frag_a(const __bf16* base, int row, int rsel) {
    v16bf a;
    const __bf16* r = base + row * KSTR;
#pragma unroll
    for (int v = 0; v < 8; ++v) {
        int kb = ((v < 4) ? 2 * v : 16 + 2 * (v - 4)) + rsel * 8;
        a[2 * v]     = r[kb];
        a[2 * v + 1] = r[kb + 1];
    }
    return a;
}
// B (32x16 bf16): lane L holds col N=L%16; v<4 -> K=2v+{0,1}(+8 if L>=16), v>=4 -> K=16+2(v-4)+{0,1}(+8 if L>=16)
__device__ __forceinline__ v16bf load_frag_b(const __bf16* base, int col, int rsel) {
    v16bf b;
    const __bf16* r = base + col * KSTR;
#pragma unroll
    for (int v = 0; v < 8; ++v) {
        int kb = ((v < 4) ? 2 * v : 16 + 2 * (v - 4)) + rsel * 8;
        b[2 * v]     = r[kb];
        b[2 * v + 1] = r[kb + 1];
    }
    return b;
}
// C/D (16x16 f32): lane L col N=L%16; VGPR v row M = v (+8 if L>=16)
__device__ __forceinline__ void store_acc(float* __restrict__ C, const v8f c, int r0, int col,
                                          int M, int N, int ldc, int relu) {
    if (col >= N) return;
#pragma unroll
    for (int v = 0; v < 8; ++v) {
        int r = r0 + v;
        if (r < M) {
            float x = c[v];
            if (relu) x = fmaxf(x, 0.0f);
            C[(size_t)r * ldc + col] = x;
        }
    }
}

// ---------------- Generic GEMM: C[M,N] = A[M,K](row major) * B[K,N](row major) ----------------
// Block tile 128x128x32; 8 waves as 4(M) x 2(N); wave tile 32x64 = 2x4 WMMA frags.
__global__ __launch_bounds__(256) void gemm_bf16_kernel(
    const float* __restrict__ A, const float* __restrict__ B, float* __restrict__ C,
    int M, int N, int K, int lda, int ldb, int ldc, int relu)
{
    __shared__ __bf16 As[BM][KSTR];
    __shared__ __bf16 Bs[BN][KSTR];
    const int tid  = threadIdx.x;
    const int m0   = blockIdx.y * BM;
    const int n0   = blockIdx.x * BN;
    const int lane = tid & 31;
    const int wv   = tid >> 5;
    const int mwb  = (wv >> 1) * 32;  // 0,32,64,96
    const int nwb  = (wv & 1) * 64;   // 0,64
    const int rsel = (lane < 16) ? 0 : 1;
    const int lr   = lane & 15;

    v8f acc[2][4];
#pragma unroll
    for (int i = 0; i < 2; ++i)
#pragma unroll
        for (int j = 0; j < 4; ++j) acc[i][j] = v8f{};

    // A staging map: row arow, 16 consecutive k
    const int arow  = tid >> 1;
    const int acol0 = (tid & 1) * 16;
    // B staging map: lane gnl over n (coalesced 128-wide), 8 k-pairs per thread
    const int gnl = tid & 127;
    const int kp2 = (tid >> 7) * 2;   // 0,2

    const bool fullM = (m0 + BM) <= M;
    const bool fullN = (n0 + BN) <= N;

    for (int k0 = 0; k0 < K; k0 += BK) {
        const bool fullK = (k0 + BK) <= K;
        {   // ---- stage A tile (f32 -> packed bf16) ----
            const int gm = m0 + arow;
            const int kb = k0 + acol0;
            if (fullM && fullK) {
                const float4* ap = (const float4*)(A + (size_t)gm * lda + kb);
                float4 x0 = ap[0], x1 = ap[1], x2 = ap[2], x3 = ap[3];
                uint4 lo, hi;
                lo.x = pack_bf16(x0.x, x0.y); lo.y = pack_bf16(x0.z, x0.w);
                lo.z = pack_bf16(x1.x, x1.y); lo.w = pack_bf16(x1.z, x1.w);
                hi.x = pack_bf16(x2.x, x2.y); hi.y = pack_bf16(x2.z, x2.w);
                hi.z = pack_bf16(x3.x, x3.y); hi.w = pack_bf16(x3.z, x3.w);
                uint4* dst = (uint4*)&As[arow][acol0];
                dst[0] = lo; dst[1] = hi;
                if (k0 + 2 * BK <= K)
                    __builtin_prefetch(A + (size_t)gm * lda + kb + BK, 0, 3);
            } else {
                const int cm = (gm < M) ? gm : (M - 1);
#pragma unroll
                for (int i = 0; i < 16; i += 2) {
                    int gk0 = kb + i, gk1 = kb + i + 1;
                    float v0 = A[(size_t)cm * lda + ((gk0 < K) ? gk0 : (K - 1))];
                    float v1 = A[(size_t)cm * lda + ((gk1 < K) ? gk1 : (K - 1))];
                    v0 = (gm < M && gk0 < K) ? v0 : 0.0f;
                    v1 = (gm < M && gk1 < K) ? v1 : 0.0f;
                    *(unsigned*)&As[arow][acol0 + i] = pack_bf16(v0, v1);
                }
            }
        }
        {   // ---- stage B tile as [n][k] ----
            const int gn = n0 + gnl;
            if (fullN && fullK) {
#pragma unroll
                for (int j = 0; j < 8; ++j) {
                    int kk = kp2 + j * 4;   // even, 0..30
                    const float* bp = B + (size_t)(k0 + kk) * ldb + gn;
                    *(unsigned*)&Bs[gnl][kk] = pack_bf16(bp[0], bp[ldb]);
                }
                if (k0 + 2 * BK <= K)
                    __builtin_prefetch(B + (size_t)(k0 + BK + kp2) * ldb + gn, 0, 3);
            } else {
                const int cn = (gn < N) ? gn : (N - 1);
#pragma unroll
                for (int j = 0; j < 8; ++j) {
                    int kk  = kp2 + j * 4;
                    int gk0 = k0 + kk, gk1 = gk0 + 1;
                    float v0 = B[(size_t)((gk0 < K) ? gk0 : (K - 1)) * ldb + cn];
                    float v1 = B[(size_t)((gk1 < K) ? gk1 : (K - 1)) * ldb + cn];
                    v0 = (gn < N && gk0 < K) ? v0 : 0.0f;
                    v1 = (gn < N && gk1 < K) ? v1 : 0.0f;
                    *(unsigned*)&Bs[gnl][kk] = pack_bf16(v0, v1);
                }
            }
        }
        __syncthreads();
        v16bf a0 = load_frag_a(&As[0][0], mwb + lr, rsel);
        v16bf a1 = load_frag_a(&As[0][0], mwb + 16 + lr, rsel);
        v16bf bf[4];
#pragma unroll
        for (int j = 0; j < 4; ++j)
            bf[j] = load_frag_b(&Bs[0][0], nwb + j * 16 + lr, rsel);
#pragma unroll
        for (int j = 0; j < 4; ++j) {
            acc[0][j] = __builtin_amdgcn_wmma_f32_16x16x32_bf16(false, a0, false, bf[j], (short)0, acc[0][j], false, false);
            acc[1][j] = __builtin_amdgcn_wmma_f32_16x16x32_bf16(false, a1, false, bf[j], (short)0, acc[1][j], false, false);
        }
        __syncthreads();
    }
    const int rtop = m0 + mwb + ((lane < 16) ? 0 : 8);
    const int c0   = n0 + nwb + lr;
#pragma unroll
    for (int i = 0; i < 2; ++i)
#pragma unroll
        for (int j = 0; j < 4; ++j)
            store_acc(C, acc[i][j], rtop + i * 16, c0 + j * 16, M, N, ldc, relu);
}

// ---------------- Conv2 as implicit-im2col GEMM (per part) ----------------
// out[p,b,co,t] = relu( sum_{ci,kw} W2[p,co,ci,kw] * H1[p,b,ci,t+kw] )
// M=1024(co), K=5120(ci*5+kw contiguous in W2), N=64*TOUT(b,t). TQ = H1 time len.
__global__ __launch_bounds__(256) void gemm_conv2_kernel(
    const float* __restrict__ W2, const float* __restrict__ H1, float* __restrict__ OUT,
    int TQ, int TOUT, int N)
{
    __shared__ __bf16 As[BM][KSTR];
    __shared__ __bf16 Bs[BN][KSTR];
    const int p = blockIdx.z;
    const float* A = W2 + (size_t)p * 1024 * 5120;
    const int K = 5120;
    const int tid  = threadIdx.x;
    const int m0   = blockIdx.y * BM;
    const int n0   = blockIdx.x * BN;
    const int lane = tid & 31;
    const int wv   = tid >> 5;
    const int mwb  = (wv >> 1) * 32;
    const int nwb  = (wv & 1) * 64;
    const int rsel = (lane < 16) ? 0 : 1;
    const int lr   = lane & 15;

    v8f acc[2][4];
#pragma unroll
    for (int i = 0; i < 2; ++i)
#pragma unroll
        for (int j = 0; j < 4; ++j) acc[i][j] = v8f{};

    const int arow  = tid >> 1;
    const int acol0 = (tid & 1) * 16;
    // B gather map: 16 elems per thread (tile 128x32)
    const int brow  = tid >> 1;          // n local 0..127
    const int bcol0 = (tid & 1) * 16;    // k sub-chunk
    const int gn = n0 + brow;
    const bool bvalid = gn < N;
    const int cgn = bvalid ? gn : (N - 1);
    const int bb = cgn / TOUT;
    const int tt = cgn - bb * TOUT;
    const float* h1r = H1 + (size_t)(p * 64 + bb) * 1024 * TQ + tt;

    for (int k0 = 0; k0 < K; k0 += BK) {
        {   // A tile: always full, vectorized
            const float4* ap = (const float4*)(A + (size_t)(m0 + arow) * 5120 + k0 + acol0);
            float4 x0 = ap[0], x1 = ap[1], x2 = ap[2], x3 = ap[3];
            uint4 lo, hi;
            lo.x = pack_bf16(x0.x, x0.y); lo.y = pack_bf16(x0.z, x0.w);
            lo.z = pack_bf16(x1.x, x1.y); lo.w = pack_bf16(x1.z, x1.w);
            hi.x = pack_bf16(x2.x, x2.y); hi.y = pack_bf16(x2.z, x2.w);
            hi.z = pack_bf16(x3.x, x3.y); hi.w = pack_bf16(x3.z, x3.w);
            uint4* dst = (uint4*)&As[arow][acol0];
            dst[0] = lo; dst[1] = hi;
            if (k0 + 2 * BK <= K)
                __builtin_prefetch(A + (size_t)(m0 + arow) * 5120 + k0 + acol0 + BK, 0, 3);
        }
        {   // B tile: im2col gather, 16 bf16 packed into two b128 LDS stores
            float v[16];
#pragma unroll
            for (int i = 0; i < 16; ++i) {
                int kk = k0 + bcol0 + i;
                int ci = kk / 5;
                int kw = kk - ci * 5;
                v[i] = h1r[(size_t)ci * TQ + kw];
            }
            uint4 lo, hi;
            if (bvalid) {
                lo.x = pack_bf16(v[0], v[1]);   lo.y = pack_bf16(v[2], v[3]);
                lo.z = pack_bf16(v[4], v[5]);   lo.w = pack_bf16(v[6], v[7]);
                hi.x = pack_bf16(v[8], v[9]);   hi.y = pack_bf16(v[10], v[11]);
                hi.z = pack_bf16(v[12], v[13]); hi.w = pack_bf16(v[14], v[15]);
            } else {
                lo.x = lo.y = lo.z = lo.w = 0u;
                hi.x = hi.y = hi.z = hi.w = 0u;
            }
            uint4* dst = (uint4*)&Bs[brow][bcol0];
            dst[0] = lo; dst[1] = hi;
        }
        __syncthreads();
        v16bf a0 = load_frag_a(&As[0][0], mwb + lr, rsel);
        v16bf a1 = load_frag_a(&As[0][0], mwb + 16 + lr, rsel);
        v16bf bf[4];
#pragma unroll
        for (int j = 0; j < 4; ++j)
            bf[j] = load_frag_b(&Bs[0][0], nwb + j * 16 + lr, rsel);
#pragma unroll
        for (int j = 0; j < 4; ++j) {
            acc[0][j] = __builtin_amdgcn_wmma_f32_16x16x32_bf16(false, a0, false, bf[j], (short)0, acc[0][j], false, false);
            acc[1][j] = __builtin_amdgcn_wmma_f32_16x16x32_bf16(false, a1, false, bf[j], (short)0, acc[1][j], false, false);
        }
        __syncthreads();
    }
    const int rtop = m0 + mwb + ((lane < 16) ? 0 : 8);
    const int cb   = n0 + nwb + lr;
#pragma unroll
    for (int j = 0; j < 4; ++j) {
        int col = cb + j * 16;
        if (col >= N) continue;
        int ob  = col / TOUT;
        int ot  = col - ob * TOUT;
#pragma unroll
        for (int i = 0; i < 2; ++i) {
#pragma unroll
            for (int v = 0; v < 8; ++v) {
                int r0 = rtop + i * 16 + v;
                OUT[((size_t)(p * 64 + ob) * 1024 + r0) * TOUT + ot] = fmaxf(acc[i][j][v], 0.0f);
            }
        }
    }
}

// ---------------- conv1 (K=66, VALU) ----------------
__global__ void conv1_kernel(const float* __restrict__ src, const float* __restrict__ w1,
                             float* __restrict__ h1, int toff, int TOUT)
{
    int id = blockIdx.x * blockDim.x + threadIdx.x;
    int total = NPART * BSZ * DMODEL * TOUT;
    if (id >= total) return;
    int t = id % TOUT;
    int o = (id / TOUT) % DMODEL;
    int b = (id / (TOUT * DMODEL)) % BSZ;
    int p = id / (TOUT * DMODEL * BSZ);
    const float* w = w1 + ((size_t)(p * DMODEL + o) * 11) * 6;
    float acc = 0.0f;
#pragma unroll
    for (int c = 0; c < 11; ++c) {
        int f = p * 11 + c;
#pragma unroll
        for (int k = 0; k < 6; ++k) {
            float x = src[(size_t)(b * IN_N + toff + t + k) * FEAT_C + f] * 1e-3f;
            acc = fmaf(x, w[c * 6 + k], acc);
        }
    }
    h1[id] = fmaxf(acc, 0.0f);
}

// srcval[b,k,f,d] = sum_l dct[d][l] * src[b, k+l, f]
__global__ void srcval_kernel(const float* __restrict__ src, float* __restrict__ srcval)
{
    int id = blockIdx.x * blockDim.x + threadIdx.x;
    if (id >= BSZ * VWIN * FEAT_C * DCTN) return;
    int d = id % DCTN;
    int f = (id / DCTN) % FEAT_C;
    int n = (id / (DCTN * FEAT_C)) % VWIN;
    int b = id / (DCTN * FEAT_C * VWIN);
    float acc = 0.0f;
    for (int l = 0; l < VLEN; ++l)
        acc = fmaf(dct_coef(d, l), src[(size_t)(b * IN_N + n + l) * FEAT_C + f], acc);
    srcval[id] = acc;
}

// dct_in[b,f,d] (first 20 of 40): frames = [40..49, 49 x25]
__global__ void dctin_kernel(const float* __restrict__ src, float* __restrict__ dctin)
{
    int id = blockIdx.x * blockDim.x + threadIdx.x;
    if (id >= BSZ * FEAT_C * DCTN) return;
    int d = id % DCTN;
    int f = (id / DCTN) % FEAT_C;
    int b = id / (DCTN * FEAT_C);
    float acc = 0.0f;
    for (int l = 0; l < VLEN; ++l) {
        int fr = (l < 10) ? (40 + l) : 49;
        acc = fmaf(dct_coef(d, l), src[(size_t)(b * IN_N + fr) * FEAT_C + f], acc);
    }
    dctin[(size_t)(b * FEAT_C + f) * (2 * DCTN) + d] = acc;
}

// score[b,p,k] = sum_d qv[p,b,d]*keyv[p,b,d,k] + 1e-15
__global__ void score_kernel(const float* __restrict__ qv, const float* __restrict__ keyv,
                             float* __restrict__ score)
{
    int id = blockIdx.x * blockDim.x + threadIdx.x;
    if (id >= BSZ * NPART * VWIN) return;
    int k = id & 15;
    int p = (id >> 4) % NPART;
    int b = (id >> 4) / NPART;
    const float* q  = qv + (size_t)(p * 64 + b) * 1024;
    const float* kv = keyv + (size_t)(p * 64 + b) * 1024 * 16 + k;
    float s = 0.0f;
    for (int d = 0; d < 1024; ++d) s = fmaf(q[d], kv[(size_t)d * 16], s);
    score[id] = s + 1e-15f;
}

__global__ void attnorm_kernel(const float* __restrict__ score, float* __restrict__ att)
{
    int id = blockIdx.x * blockDim.x + threadIdx.x;
    if (id >= BSZ * NPART) return;
    const float* s = score + id * 16;
    float sum = 0.0f;
    for (int k = 0; k < 16; ++k) sum += s[k];
    float inv = 1.0f / sum;
    for (int k = 0; k < 16; ++k) att[id * 16 + k] = s[k] * inv;
}

// dct_in[b,f,20+d] = sum_k att[b, f/11, k] * srcval[b,k,f,d]
__global__ void dctatt_kernel(const float* __restrict__ att, const float* __restrict__ srcval,
                              float* __restrict__ dctin)
{
    int id = blockIdx.x * blockDim.x + threadIdx.x;
    if (id >= BSZ * FEAT_C * DCTN) return;
    int d = id % DCTN;
    int f = (id / DCTN) % FEAT_C;
    int b = id / (DCTN * FEAT_C);
    int p = f / 11;
    float acc = 0.0f;
    for (int k = 0; k < VWIN; ++k)
        acc = fmaf(att[(size_t)(b * NPART + p) * 16 + k],
                   srcval[((size_t)(b * VWIN + k) * FEAT_C + f) * DCTN + d], acc);
    dctin[(size_t)(b * FEAT_C + f) * (2 * DCTN) + DCTN + d] = acc;
}

// ---------------- 66x66 joint mixing + bias + (BN) + (tanh) + (residual) ----------------
__global__ __launch_bounds__(256) void attmix_kernel(
    const float* __restrict__ t, const float* __restrict__ attm, const float* __restrict__ bias,
    const float* __restrict__ g, const float* __restrict__ be, const float* __restrict__ res,
    float* __restrict__ out, int F, int do_tanh)
{
    __shared__ float attS[FEAT_C * FEAT_C];
    __shared__ float tS[FEAT_C * 64];
    const int b  = blockIdx.y;
    const int f0 = blockIdx.x * 64;
    const int tid = threadIdx.x;
    for (int i = tid; i < FEAT_C * FEAT_C; i += 256) attS[i] = attm[i];
    for (int i = tid; i < FEAT_C * 64; i += 256) {
        int m = i >> 6, fl = i & 63;
        int f = f0 + fl;
        tS[i] = (f < F) ? t[(size_t)(b * FEAT_C + m) * F + f] : 0.0f;
    }
    __syncthreads();
    const float bninv = 0.999995000037f;  // 1/sqrt(1+1e-5)
    for (int i = tid; i < FEAT_C * 64; i += 256) {
        int n = i >> 6, fl = i & 63;
        int f = f0 + fl;
        if (f >= F) continue;
        float acc = 0.0f;
        for (int m = 0; m < FEAT_C; ++m) acc = fmaf(attS[n * FEAT_C + m], tS[(m << 6) + fl], acc);
        acc += bias[f];
        if (g) acc = fmaf(acc, g[(size_t)n * F + f] * bninv, be[(size_t)n * F + f]);
        if (do_tanh) acc = tanhf(acc);
        size_t oi = (size_t)(b * FEAT_C + n) * F + f;
        if (res) acc += res[oi];
        out[oi] = acc;
    }
}

// out[b,l,0,f] = sum_d idct[l,d]*dctout[b,f,d]  (idct = dct^T, orthonormal)
__global__ void idct_out_kernel(const float* __restrict__ dctout, float* __restrict__ out)
{
    int id = blockIdx.x * blockDim.x + threadIdx.x;
    if (id >= BSZ * VLEN * FEAT_C) return;
    int f = id % FEAT_C;
    int l = (id / FEAT_C) % VLEN;
    int b = id / (FEAT_C * VLEN);
    float acc = 0.0f;
    for (int d = 0; d < DCTN; ++d)
        acc = fmaf(dct_coef(d, l), dctout[(size_t)(b * FEAT_C + f) * (2 * DCTN) + d], acc);
    out[id] = acc;
}

// ---------------- host launcher ----------------
extern "C" void kernel_launch(void* const* d_in, const int* in_sizes, int n_in,
                              void* d_out, int out_size, void* d_ws, size_t ws_size,
                              hipStream_t stream) {
    const float* src     = (const float*)d_in[0];
    const float* cQ1     = (const float*)d_in[1];
    const float* cQ2     = (const float*)d_in[2];
    const float* cK1     = (const float*)d_in[3];
    const float* cK2     = (const float*)d_in[4];
    const float* gc1_w   = (const float*)d_in[5];
    const float* gc1_att = (const float*)d_in[6];
    const float* gc1_b   = (const float*)d_in[7];
    const float* bn1_g   = (const float*)d_in[8];
    const float* bn1_b   = (const float*)d_in[9];
    const float* bw1     = (const float*)d_in[10];
    const float* ba1     = (const float*)d_in[11];
    const float* bb1     = (const float*)d_in[12];
    const float* bg1     = (const float*)d_in[13];
    const float* bbe1    = (const float*)d_in[14];
    const float* bw2     = (const float*)d_in[15];
    const float* ba2     = (const float*)d_in[16];
    const float* bb2     = (const float*)d_in[17];
    const float* bg2     = (const float*)d_in[18];
    const float* bbe2    = (const float*)d_in[19];
    const float* gc7_w   = (const float*)d_in[20];
    const float* gc7_att = (const float*)d_in[21];
    const float* gc7_b   = (const float*)d_in[22];
    float* out = (float*)d_out;
    float* ws  = (float*)d_ws;

    // workspace layout (floats)
    float* SRCVAL = ws;                      // 64*16*66*20 = 1,351,680
    float* H1K    = SRCVAL + 1351680;        // 6*64*1024*20 = 7,864,320
    float* H1Q    = H1K    + 7864320;        // 6*64*1024*5  = 1,966,080
    float* KEYV   = H1Q    + 1966080;        // 6*64*1024*16 = 6,291,456
    float* QV     = KEYV   + 6291456;        // 6*64*1024    =   393,216
    float* SCORE  = QV     + 393216;         // 64*6*16
    float* ATT    = SCORE  + 6144;           // 64*6*16
    float* DCTIN  = ATT    + 6144;           // 4224*40
    float* TBUF   = DCTIN  + 168960;         // 4224*1024
    float* YBUF   = TBUF   + 4325376;        // 4224*1024
    float* Y1BUF  = YBUF   + 4325376;        // 4224*1024
    float* DCTOUT = Y1BUF  + 4325376;        // 4224*40

    const int T = 256;
    srcval_kernel<<<(1351680 + T - 1) / T, T, 0, stream>>>(src, SRCVAL);
    dctin_kernel<<<(84480 + T - 1) / T, T, 0, stream>>>(src, DCTIN);
    conv1_kernel<<<(6 * 64 * 1024 * 20 + T - 1) / T, T, 0, stream>>>(src, cK1, H1K, 0, 20);
    conv1_kernel<<<(6 * 64 * 1024 * 5 + T - 1) / T, T, 0, stream>>>(src, cQ1, H1Q, 40, 5);
    gemm_conv2_kernel<<<dim3(8, 8, 6), T, 0, stream>>>(cK2, H1K, KEYV, 20, 16, 1024);
    gemm_conv2_kernel<<<dim3(1, 8, 6), T, 0, stream>>>(cQ2, H1Q, QV, 5, 1, 64);
    score_kernel<<<(6144 + T - 1) / T, T, 0, stream>>>(QV, KEYV, SCORE);
    attnorm_kernel<<<(384 + 63) / 64, 64, 0, stream>>>(SCORE, ATT);
    dctatt_kernel<<<(84480 + T - 1) / T, T, 0, stream>>>(ATT, SRCVAL, DCTIN);

    gemm_bf16_kernel<<<dim3(8, 33), T, 0, stream>>>(DCTIN, gc1_w, TBUF,
                                                    4224, 1024, 40, 40, 1024, 1024, 0);
    attmix_kernel<<<dim3(16, 64), T, 0, stream>>>(TBUF, gc1_att, gc1_b, bn1_g, bn1_b,
                                                  nullptr, YBUF, 1024, 1);
    for (int s = 0; s < 2; ++s) {
        const float* w1  = bw1 + (size_t)s * 1024 * 1024;
        const float* w2  = bw2 + (size_t)s * 1024 * 1024;
        const float* a1  = ba1 + (size_t)s * 66 * 66;
        const float* a2  = ba2 + (size_t)s * 66 * 66;
        const float* b1  = bb1 + (size_t)s * 1024;
        const float* b2  = bb2 + (size_t)s * 1024;
        const float* g1  = bg1 + (size_t)s * 66 * 1024;
        const float* be1 = bbe1 + (size_t)s * 66 * 1024;
        const float* g2  = bg2 + (size_t)s * 66 * 1024;
        const float* be2 = bbe2 + (size_t)s * 66 * 1024;
        gemm_bf16_kernel<<<dim3(8, 33), T, 0, stream>>>(YBUF, w1, TBUF,
                                                        4224, 1024, 1024, 1024, 1024, 1024, 0);
        attmix_kernel<<<dim3(16, 64), T, 0, stream>>>(TBUF, a1, b1, g1, be1, nullptr, Y1BUF, 1024, 1);
        gemm_bf16_kernel<<<dim3(8, 33), T, 0, stream>>>(Y1BUF, w2, TBUF,
                                                        4224, 1024, 1024, 1024, 1024, 1024, 0);
        attmix_kernel<<<dim3(16, 64), T, 0, stream>>>(TBUF, a2, b2, g2, be2, YBUF, YBUF, 1024, 1);
    }
    gemm_bf16_kernel<<<dim3(1, 33), T, 0, stream>>>(YBUF, gc7_w, TBUF,
                                                    4224, 40, 1024, 1024, 40, 40, 0);
    attmix_kernel<<<dim3(1, 64), T, 0, stream>>>(TBUF, gc7_att, gc7_b, nullptr, nullptr,
                                                 DCTIN, DCTOUT, 40, 0);
    idct_out_kernel<<<(147840 + T - 1) / T, T, 0, stream>>>(DCTOUT, out);
}
